// AttentionHead_66151086293237
// MI455X (gfx1250) — matrix operand order repaired and checked by reading, not compiled, not run on hardware
//
#include <hip/hip_runtime.h>

// ---------------------------------------------------------------------------
// Single-head attention for MI455X (gfx1250, wave32, WMMA + TDM).
// qkv = x@Wqkv+b ; scores=qk^T/8 ; softmax ; ctx=attn@v ; out=ctx@Wout+b
// All GEMMs via v_wmma_f32_16x16x32_bf16 (f32 accumulate).  Flash-style
// streaming softmax (256MB score tensor never touches HBM).  Shared operand
// panels (W, K, V tiles) are staged into LDS with the Tensor Data Mover
// (tensor_load_to_lds), double-buffered, synced with s_wait_tensorcnt.
// ---------------------------------------------------------------------------

typedef __attribute__((ext_vector_type(16))) __bf16 bf16x16;
typedef __attribute__((ext_vector_type(8)))  __bf16 bf16x8;
typedef __attribute__((ext_vector_type(8)))  float  f32x8;
typedef __attribute__((ext_vector_type(4)))  unsigned u32x4;
typedef __attribute__((ext_vector_type(8)))  int      i32x8;
typedef __attribute__((ext_vector_type(4)))  int      i32x4;

#define EMB   1024
#define HEAD  64
#define NQKV  192
#define SEQ   4096
#define NBATCH 4
#define ROWS  (NBATCH * SEQ)   // 16384

__device__ __forceinline__ __bf16 to_bf(float f) { return (__bf16)f; }

__device__ __forceinline__ f32x8 zero8() {
    f32x8 v = {0.f,0.f,0.f,0.f,0.f,0.f,0.f,0.f};
    return v;
}

__device__ __forceinline__ f32x8 wmma_bf16(bf16x16 a, bf16x16 b, f32x8 c) {
    return __builtin_amdgcn_wmma_f32_16x16x32_bf16(
        false, a, false, b, (short)0, c, false, false);
}

// Flat shared-pointer low 32 bits == LDS allocation-relative byte offset.
__device__ __forceinline__ unsigned lds_off(const void* p) {
    return (unsigned)(unsigned long long)(uintptr_t)p;
}

// ---------------------------------------------------------------------------
// TDM: load a 2-D tile of bf16 (data_size = 2B) from global into LDS.
//   tile    : tile_d0 (contiguous elems) x tile_d1 (rows)
//   tensor  : tensor_d0 x tensor_d1, row stride d0_stride (elements)
// D# per CDNA5 ISA.  Groups 2/3 zero (2-D tile),
// 6-arg builtin form (this toolchain): (g0, g1, g2, g3, g_extra, cpol).
// ---------------------------------------------------------------------------
__device__ __forceinline__ void tdm_load_2d(unsigned ldsoff, const void* gptr,
                                            unsigned tile_d0, unsigned tile_d1,
                                            unsigned tensor_d0, unsigned tensor_d1,
                                            unsigned long long d0_stride) {
    unsigned long long ga = (unsigned long long)(uintptr_t)gptr;
    u32x4 g0;
    g0[0] = 1u;                                        // count=1, user descriptor
    g0[1] = ldsoff;                                    // lds_addr (bytes)
    g0[2] = (unsigned)(ga & 0xFFFFFFFFu);              // global_addr[31:0]
    g0[3] = (unsigned)((ga >> 32) & 0x1FFFFFFu)        // global_addr[56:32]
          | (2u << 30);                                // type=2 ("image")
    i32x8 g1;
    g1[0] = (int)(1u << 16);                           // data_size=1 (2 bytes)
    g1[1] = (int)((tensor_d0 & 0xFFFFu) << 16);        // tensor_dim0[15:0]
    g1[2] = (int)(((tensor_d0 >> 16) & 0xFFFFu)        // tensor_dim0[31:16]
          |        ((tensor_d1 & 0xFFFFu) << 16));     // tensor_dim1[15:0]
    g1[3] = (int)(((tensor_d1 >> 16) & 0xFFFFu)        // tensor_dim1[31:16]
          |        ((tile_d0 & 0xFFFFu) << 16));       // tile_dim0
    g1[4] = (int)(tile_d1 & 0xFFFFu);                  // tile_dim1 (tile_dim2=0)
    g1[5] = (int)(unsigned)(d0_stride & 0xFFFFFFFFull);// dim0_stride[31:0]
    g1[6] = (int)(unsigned)((d0_stride >> 32) & 0xFFFFull); // [47:32]
    g1[7] = 0;
    i32x4 z4 = {0, 0, 0, 0};
    i32x8 z8 = {0, 0, 0, 0, 0, 0, 0, 0};
    __builtin_amdgcn_tensor_load_to_lds(g0, g1, z4, z4, z8, 0);
}

// ---------------------------------------------------------------------------
// Kernel 0: transpose + convert weights to bf16 so B-fragments are contiguous.
//   WqT[n][k] = Wqkv[k][n]   (NQKV x EMB)
//   WoT[e][h] = Wout[h][e]   (EMB  x HEAD)
// ---------------------------------------------------------------------------
__global__ void prep_weights(const float* __restrict__ Wqkv,
                             const float* __restrict__ Wout,
                             __bf16* __restrict__ WqT,
                             __bf16* __restrict__ WoT) {
    int idx = blockIdx.x * 256 + threadIdx.x;
    if (idx < NQKV * EMB) {
        int n = idx / EMB, k = idx % EMB;
        WqT[idx] = to_bf(Wqkv[k * NQKV + n]);
    }
    if (idx < EMB * HEAD) {
        int e = idx / HEAD, h = idx % HEAD;
        WoT[idx] = to_bf(Wout[h * EMB + e]);
    }
}

// ---------------------------------------------------------------------------
// Kernel 1: qkv projection.  M=16384 rows, N=192, K=1024.
// Wave = 16 rows x 192 cols (12 accumulators).  The shared 32x192 W panel of
// each K-step is TDM-loaded into LDS (double-buffered) by wave 0; all 8 waves
// feed WMMA B-fragments from LDS.  Writes q,k row-major [rows][64] bf16 and
// v transposed [B][64][SEQ] bf16.
// ---------------------------------------------------------------------------
__global__ __launch_bounds__(256)
void qkv_proj(const float* __restrict__ x,
              const __bf16* __restrict__ WqT,
              const float* __restrict__ bqkv,
              __bf16* __restrict__ q,
              __bf16* __restrict__ kmat,
              __bf16* __restrict__ vT) {
    __shared__ alignas(16) __bf16 Wbuf[2][NQKV][32];   // 2 x 12 KB

    const int lane = threadIdx.x & 31;
    const int wave = threadIdx.x >> 5;
    const int h = lane >> 4;
    const int r = lane & 15;
    const int m0 = blockIdx.x * 128 + wave * 16;

    f32x8 acc[12];
#pragma unroll
    for (int t = 0; t < 12; ++t) acc[t] = zero8();

    const float* xrow = x + (size_t)(m0 + r) * EMB;

    if (wave == 0)
        tdm_load_2d(lds_off(&Wbuf[0][0][0]), WqT, 32, NQKV, EMB, NQKV, EMB);

    for (int ks = 0; ks < 32; ++ks) {
        const int p = ks & 1;
        if (wave == 0) {
            if (ks + 1 < 32) {
                tdm_load_2d(lds_off(&Wbuf[1 - p][0][0]), WqT + (ks + 1) * 32,
                            32, NQKV, EMB, NQKV, EMB);
                __builtin_amdgcn_s_wait_tensorcnt(1);  // buf[p] done (in-order)
            } else {
                __builtin_amdgcn_s_wait_tensorcnt(0);
            }
        }
        __syncthreads();                       // Wbuf[p] ready for everyone

        // A fragment (16x32): lane row r, K = h*16+{0..15}
        union { bf16x16 v; __bf16 e[16]; } A;
        const float* pa0 = xrow + ks * 32 + h * 16;
        const float* pa1 = xrow + ks * 32 + h * 16 + 8;
#pragma unroll
        for (int i = 0; i < 8; ++i) {
            A.e[i]     = to_bf(pa0[i]);
            A.e[8 + i] = to_bf(pa1[i]);
        }
#pragma unroll
        for (int t = 0; t < 12; ++t) {
            // B fragment (32x16): lane = column (t*16+r), K = h*16+{0..15}
            bf16x16 B = *(const bf16x16*)&Wbuf[p][t * 16 + r][h * 16];
            acc[t] = wmma_bf16(A.v, B, acc[t]);
        }
        __syncthreads();                       // done reading Wbuf[p]
    }

    // C layout: VGPR i -> row m0 + i + h*8, col = r (within tile)
#pragma unroll
    for (int t = 0; t < 12; ++t) {
        const int n = t * 16 + r;
        const float bias = bqkv[n];
#pragma unroll
        for (int i = 0; i < 8; ++i) {
            const int row = m0 + i + h * 8;
            const float val = acc[t][i] + bias;
            if (t < 4) {
                q[(size_t)row * HEAD + n] = to_bf(val);
            } else if (t < 8) {
                kmat[(size_t)row * HEAD + (n - 64)] = to_bf(val);
            } else {
                const int b = row >> 12, s = row & (SEQ - 1);
                vT[((size_t)(b * HEAD) + (n - 128)) * SEQ + s] = to_bf(val);
            }
        }
    }
}

// ---------------------------------------------------------------------------
// Kernel 2: flash attention.  Wave = 16 queries; the workgroup's 8 waves share
// each 32-key K tile (32x64) and V^T tile (64x32), TDM-loaded into LDS and
// double-buffered.  Softmax stats live in the WMMA C layout (row stats per
// VGPR index, reduced across 16-lane halves with shfl_xor).  P tile is
// redistributed C-layout -> A-layout through wave-private LDS.
// ---------------------------------------------------------------------------
__global__ __launch_bounds__(256)
void flash_attn(const __bf16* __restrict__ q,
                const __bf16* __restrict__ kmat,
                const __bf16* __restrict__ vT,
                __bf16* __restrict__ ctx) {
    __shared__ alignas(16) __bf16 Kbuf[2][32][HEAD];   // 2 x 4 KB
    __shared__ alignas(16) __bf16 Vbuf[2][HEAD][32];   // 2 x 4 KB
    __shared__ alignas(16) __bf16 Pbuf[8][16][32];     // 8 KB (per-wave P tile)

    const int lane = threadIdx.x & 31;
    const int wave = threadIdx.x >> 5;
    const int h = lane >> 4;
    const int r = lane & 15;
    const int q0 = blockIdx.x * 128 + wave * 16;       // global query row
    const int b  = q0 >> 12;                           // batch (4096 rows each)
    const size_t kbase = (size_t)b * SEQ;

    // Q fragments for K=64 (two 16x32 A-frags), loaded once.
    bf16x16 qa[2];
    {
        const __bf16* qrow = q + (size_t)(q0 + r) * HEAD;
#pragma unroll
        for (int s = 0; s < 2; ++s) {
            union { bf16x16 v; bf16x8 p[2]; } u;
            u.p[0] = *(const bf16x8*)(qrow + s * 32 + h * 16);
            u.p[1] = *(const bf16x8*)(qrow + s * 32 + h * 16 + 8);
            qa[s] = u.v;
        }
    }

    const float scale = 0.125f;   // 1/sqrt(64)
    float m_i[8], l_i[8];
#pragma unroll
    for (int i = 0; i < 8; ++i) { m_i[i] = -3.0e38f; l_i[i] = 0.f; }
    f32x8 acc[4];
#pragma unroll
    for (int t = 0; t < 4; ++t) acc[t] = zero8();

    const int nblk = SEQ / 32;                          // 128
    if (wave == 0) {
        tdm_load_2d(lds_off(&Kbuf[0][0][0]), kmat + kbase * HEAD,
                    HEAD, 32, HEAD, ROWS, HEAD);
        tdm_load_2d(lds_off(&Vbuf[0][0][0]), vT + (size_t)(b * HEAD) * SEQ,
                    32, HEAD, SEQ, NBATCH * HEAD, SEQ);
    }

    for (int blk = 0; blk < nblk; ++blk) {
        const int p = blk & 1;
        const int key0 = blk * 32;
        if (wave == 0) {
            if (blk + 1 < nblk) {
                const int kn = key0 + 32;
                tdm_load_2d(lds_off(&Kbuf[1 - p][0][0]),
                            kmat + (kbase + kn) * HEAD,
                            HEAD, 32, HEAD, ROWS, HEAD);
                tdm_load_2d(lds_off(&Vbuf[1 - p][0][0]),
                            vT + (size_t)(b * HEAD) * SEQ + kn,
                            32, HEAD, SEQ, NBATCH * HEAD, SEQ);
                __builtin_amdgcn_s_wait_tensorcnt(2);  // pair p done (in-order)
            } else {
                __builtin_amdgcn_s_wait_tensorcnt(0);
            }
        }
        __syncthreads();                    // K/V buf[p] ready for everyone

        // ---- scores: 2 key tiles x 2 K-steps = 4 WMMA --------------------
        f32x8 s[2];
#pragma unroll
        for (int t = 0; t < 2; ++t) {
            f32x8 sc = zero8();
#pragma unroll
            for (int kk = 0; kk < 2; ++kk) {
                // B frag: lane=key col (t*16+r), K = head dims kk*32+h*16+..
                bf16x16 B = *(const bf16x16*)&Kbuf[p][t * 16 + r][kk * 32 + h * 16];
                sc = wmma_bf16(qa[kk], B, sc);
            }
            s[t] = sc;
        }
#pragma unroll
        for (int t = 0; t < 2; ++t)
#pragma unroll
            for (int i = 0; i < 8; ++i) s[t][i] *= scale;

        // ---- online softmax ----------------------------------------------
        float alpha[8];
#pragma unroll
        for (int i = 0; i < 8; ++i) {
            float mx = fmaxf(s[0][i], s[1][i]);
#pragma unroll
            for (int off = 8; off >= 1; off >>= 1)
                mx = fmaxf(mx, __shfl_xor(mx, off, 32));
            const float mnew = fmaxf(m_i[i], mx);
            alpha[i] = __expf(m_i[i] - mnew);
            m_i[i] = mnew;
        }
#pragma unroll
        for (int i = 0; i < 8; ++i) {
            s[0][i] = __expf(s[0][i] - m_i[i]);
            s[1][i] = __expf(s[1][i] - m_i[i]);
            float rsum = s[0][i] + s[1][i];
#pragma unroll
            for (int off = 8; off >= 1; off >>= 1)
                rsum += __shfl_xor(rsum, off, 32);
            l_i[i] = l_i[i] * alpha[i] + rsum;
#pragma unroll
            for (int t = 0; t < 4; ++t) acc[t][i] *= alpha[i];
        }

        // ---- P: C layout -> LDS -> A layout (wave-private) ---------------
#pragma unroll
        for (int t = 0; t < 2; ++t)
#pragma unroll
            for (int i = 0; i < 8; ++i)
                Pbuf[wave][i + h * 8][t * 16 + r] = to_bf(s[t][i]);

        asm volatile("s_wait_dscnt 0" ::: "memory");   // wave-sync LDS RAW

        union { bf16x16 v; bf16x8 pp[2]; } P;
        P.pp[0] = *(const bf16x8*)&Pbuf[wave][r][h * 16];
        P.pp[1] = *(const bf16x8*)&Pbuf[wave][r][h * 16 + 8];

        // ---- acc += P @ V : 4 WMMA (head-dim tiles) ----------------------
#pragma unroll
        for (int t = 0; t < 4; ++t) {
            // B frag: lane = head-dim col (t*16+r), K = keys h*16+{0..15}
            bf16x16 Bv = *(const bf16x16*)&Vbuf[p][t * 16 + r][h * 16];
            acc[t] = wmma_bf16(P.v, Bv, acc[t]);
        }
        __syncthreads();                    // done reading K/V buf[p]
    }

    // ---- normalize + store ctx (bf16 row-major [rows][64]) ---------------
#pragma unroll
    for (int t = 0; t < 4; ++t) {
#pragma unroll
        for (int i = 0; i < 8; ++i) {
            const int row = q0 + i + h * 8;
            const float val = acc[t][i] / l_i[i];
            ctx[(size_t)row * HEAD + t * 16 + r] = to_bf(val);
        }
    }
}

// ---------------------------------------------------------------------------
// Kernel 3: out projection.  M=16384, N=1024, K=64.  Wave = 16x64 tile.
// Grid: (128 M-blocks, 16 N-blocks).  W panel is tiny (128KB bf16) and
// N-partitioned across blocks -> plain global loads hit L2.
// ---------------------------------------------------------------------------
__global__ __launch_bounds__(256)
void out_proj(const __bf16* __restrict__ ctx,
              const __bf16* __restrict__ WoT,
              const float* __restrict__ bout,
              float* __restrict__ out) {
    const int lane = threadIdx.x & 31;
    const int wave = threadIdx.x >> 5;
    const int h = lane >> 4;
    const int r = lane & 15;
    const int m0 = blockIdx.x * 128 + wave * 16;
    const int n0 = blockIdx.y * 64;

    bf16x16 a[2];
    {
        const __bf16* crow = ctx + (size_t)(m0 + r) * HEAD;
#pragma unroll
        for (int s = 0; s < 2; ++s) {
            union { bf16x16 v; bf16x8 p[2]; } u;
            u.p[0] = *(const bf16x8*)(crow + s * 32 + h * 16);
            u.p[1] = *(const bf16x8*)(crow + s * 32 + h * 16 + 8);
            a[s] = u.v;
        }
    }
#pragma unroll
    for (int t = 0; t < 4; ++t) {
        f32x8 acc = zero8();
        const __bf16* wrow = WoT + (size_t)(n0 + t * 16 + r) * HEAD;
        acc = wmma_bf16(a[0], *(const bf16x16*)(wrow + h * 16), acc);
        acc = wmma_bf16(a[1], *(const bf16x16*)(wrow + 32 + h * 16), acc);
        const float bias = bout[n0 + t * 16 + r];
#pragma unroll
        for (int i = 0; i < 8; ++i)
            out[(size_t)(m0 + i + h * 8) * EMB + n0 + t * 16 + r] = acc[i] + bias;
    }
}

// ---------------------------------------------------------------------------
extern "C" void kernel_launch(void* const* d_in, const int* in_sizes, int n_in,
                              void* d_out, int out_size, void* d_ws, size_t ws_size,
                              hipStream_t stream) {
    const float* x    = (const float*)d_in[0];
    const float* Wqkv = (const float*)d_in[1];
    const float* bqkv = (const float*)d_in[2];
    const float* Wout = (const float*)d_in[3];
    const float* bout = (const float*)d_in[4];
    float* out = (float*)d_out;

    // workspace layout (~8.9 MB total, all offsets 128B-aligned)
    char* ws = (char*)d_ws;
    __bf16* WqT = (__bf16*)ws;  ws += (size_t)NQKV * EMB * 2;   // 384 KB
    __bf16* WoT = (__bf16*)ws;  ws += (size_t)EMB * HEAD * 2;   // 128 KB
    __bf16* q   = (__bf16*)ws;  ws += (size_t)ROWS * HEAD * 2;  // 2 MB
    __bf16* k   = (__bf16*)ws;  ws += (size_t)ROWS * HEAD * 2;  // 2 MB
    __bf16* vT  = (__bf16*)ws;  ws += (size_t)ROWS * HEAD * 2;  // 2 MB
    __bf16* ctx = (__bf16*)ws;                                  // 2 MB

    prep_weights<<<768, 256, 0, stream>>>(Wqkv, Wout, WqT, WoT);
    qkv_proj<<<ROWS / 128, 256, 0, stream>>>(x, WqT, bqkv, q, k, vT);
    flash_attn<<<ROWS / 128, 256, 0, stream>>>(q, k, vT, ctx);
    out_proj<<<dim3(ROWS / 128, EMB / 64), 256, 0, stream>>>(ctx, WoT, bout, out);
}
